// ShapegraphEncoder_18219251269655
// MI455X (gfx1250) — matrix operand order repaired and checked by reading, not compiled, not run on hardware
//
#include <hip/hip_runtime.h>

// ---------------- problem constants (from reference) ----------------
#define NNODES   20000
#define NEDGES   320000
#define NB       64
#define NODE_DIM 128
#define HID      256
#define NHEADS   4
#define NLAYERS  3
#define EMB      256
#define CDIM     64                       // HID / HEADS
#define MW       (NB + NNODES)            // 20064 augmented nodes
#define EAUG     (NEDGES + 2*NNODES + MW) // 380064 augmented edges
#define NEG_SLOPE 0.2f
#define LN_EPS    1e-5f

typedef __attribute__((ext_vector_type(16))) _Float16 v16h;
typedef __attribute__((ext_vector_type(8)))  _Float16 v8h;
typedef __attribute__((ext_vector_type(8)))  float    v8f;

#define WMMA_F16(A, B, C) __builtin_amdgcn_wmma_f32_16x16x32_f16(false, (A), false, (B), (short)0, (C), false, false)

// ---------------- helpers ----------------
__device__ inline float atomicMaxF(float* addr, float val) {
    unsigned int* ua = (unsigned int*)addr;
    unsigned int old = *ua, assumed;
    do {
        assumed = old;
        if (__uint_as_float(assumed) >= val) break;
        old = atomicCAS(ua, assumed, __float_as_uint(val));
    } while (old != assumed);
    return __uint_as_float(old);
}

// Decode augmented edge list (reference: shifted edges, cls->node, node->cls, self loops)
__device__ inline void edge_sd(int e, const int* __restrict__ ei,
                               const int* __restrict__ batch, int& s, int& d) {
    if (e < NEDGES)                { s = ei[e] + NB;            d = ei[NEDGES + e] + NB; }
    else if (e < NEDGES + NNODES)  { int i = e - NEDGES;        s = batch[i]; d = i + NB; }
    else if (e < NEDGES + 2*NNODES){ int i = e - NEDGES-NNODES; s = i + NB;   d = batch[i]; }
    else                           { int i = e - NEDGES-2*NNODES; s = i; d = i; }
}

// f32 -> f16 conversion (done once per operand instead of per GEMM block)
__global__ void k_cvt_f16(const float* __restrict__ src, _Float16* __restrict__ dst, int n) {
    int i = blockIdx.x * blockDim.x + threadIdx.x;
    if (i < n) dst[i] = (_Float16)src[i];
}

// ---------------- WMMA GEMM: Y[M,256] = X16[M,K] @ W16[K,256] + bias ----------------
// Block = 128 threads (4 waves) computing a 32-row x 256-col output tile.
// A tile (32 x K f16) staged once in LDS, shared by all waves. Wave w owns cols
// [w*64, w*64+64) with 8 accumulators (2 row-tiles x 4 col-tiles): 8 WMMAs per
// 32-wide K step reusing one set of B fragments. No f32->f16 cvt in the hot loop.
// Requires: M % 32 == 0, NOUT == 256, K in {128,256} (compile-time, full unroll).
// Fragment layouts per CDNA5 ISA 7.12.2 (wave32):
//   A 16x32: lanes 0-15 hold M=lane, K={0..7,16..23}; lanes 16-31 hold M=lane-16, K={8..15,24..31}
//   B 32x16: lane l holds N=l&15; K as for A (lanes 0-15 K={0..7,16..23}, 16-31 K={8..15,24..31})
//   C/D:     lane -> N=lane&15, VGPR r -> M = r + 8*(lane>=16)
template<int K>
__global__ __launch_bounds__(128)
void k_wmma_gemm(const _Float16* __restrict__ X, const _Float16* __restrict__ W,
                 const float* __restrict__ bias, float* __restrict__ Y)
{
    constexpr int KP = K + 8;                  // row pad (16B multiple, bank-staggered)
    __shared__ __align__(16) _Float16 As[32][KP];

    const int tid  = threadIdx.x;
    const int wave = tid >> 5;
    const int lane = tid & 31;
    const int kh   = lane >> 4;                // 0: low K group, 1: high K group
    const int l15  = lane & 15;
    const int row0 = blockIdx.x * 32;
    const int col0 = wave * 64;

    // stage A tile: 32 rows x K f16, 16B chunks, coalesced
    constexpr int CH = 32 * K / 8;             // 8 halves per chunk
    #pragma unroll
    for (int c = tid; c < CH; c += 128) {
        const int r = c / (K / 8);
        const int k = (c % (K / 8)) * 8;
        *(v8h*)&As[r][k] = *(const v8h*)(X + (size_t)(row0 + r) * K + k);
    }
    __builtin_prefetch(W + (size_t)lane * 256 + col0, 0, 3);   // -> global_prefetch_b8
    __syncthreads();

    v8f acc0 = {}, acc1 = {}, acc2 = {}, acc3 = {};
    v8f acc4 = {}, acc5 = {}, acc6 = {}, acc7 = {};
    const _Float16* __restrict__ arow0 = &As[l15][0];
    const _Float16* __restrict__ arow1 = &As[16 + l15][0];

    #pragma unroll
    for (int k0 = 0; k0 < K; k0 += 32) {
        const int ka = k0 + kh * 8;
        // A fragments from LDS (ds_load_b128 x4)
        v16h a0, a1;
        {
            const v8h lo = *(const v8h*)(arow0 + ka);
            const v8h hi = *(const v8h*)(arow0 + ka + 16);
            #pragma unroll
            for (int j = 0; j < 8; ++j) { a0[j] = lo[j]; a0[8 + j] = hi[j]; }
        }
        {
            const v8h lo = *(const v8h*)(arow1 + ka);
            const v8h hi = *(const v8h*)(arow1 + ka + 16);
            #pragma unroll
            for (int j = 0; j < 8; ++j) { a1[j] = lo[j]; a1[8 + j] = hi[j]; }
        }
        // B fragments: lane l holds col (l&15) of each tile, K = ka+{0..7}, ka+16+{0..7}
        const _Float16* __restrict__ wrow = W + (size_t)ka * 256 + col0 + l15;
        v16h b0, b1, b2, b3;
        #pragma unroll
        for (int j = 0; j < 8; ++j) {
            b0[j] = wrow[j * 256];      b0[8 + j] = wrow[(16 + j) * 256];
            b1[j] = wrow[j * 256 + 16]; b1[8 + j] = wrow[(16 + j) * 256 + 16];
            b2[j] = wrow[j * 256 + 32]; b2[8 + j] = wrow[(16 + j) * 256 + 32];
            b3[j] = wrow[j * 256 + 48]; b3[8 + j] = wrow[(16 + j) * 256 + 48];
        }

        acc0 = WMMA_F16(a0, b0, acc0);
        acc1 = WMMA_F16(a0, b1, acc1);
        acc2 = WMMA_F16(a0, b2, acc2);
        acc3 = WMMA_F16(a0, b3, acc3);
        acc4 = WMMA_F16(a1, b0, acc4);
        acc5 = WMMA_F16(a1, b1, acc5);
        acc6 = WMMA_F16(a1, b2, acc6);
        acc7 = WMMA_F16(a1, b3, acc7);
    }

    // epilogue: straight-line stores (M % 32 == 0, stride 256)
    float bv0 = 0.f, bv1 = 0.f, bv2 = 0.f, bv3 = 0.f;
    if (bias) {
        bv0 = bias[col0 + l15];      bv1 = bias[col0 + 16 + l15];
        bv2 = bias[col0 + 32 + l15]; bv3 = bias[col0 + 48 + l15];
    }
    #pragma unroll
    for (int r = 0; r < 8; ++r) {
        float* __restrict__ y0 = Y + (size_t)(row0 + r + kh * 8) * 256 + col0 + l15;
        y0[0]  = acc0[r] + bv0;
        y0[16] = acc1[r] + bv1;
        y0[32] = acc2[r] + bv2;
        y0[48] = acc3[r] + bv3;
        float* __restrict__ y1 = y0 + 16 * 256;          // rows 16..31 of the tile
        y1[0]  = acc4[r] + bv0;
        y1[16] = acc5[r] + bv1;
        y1[32] = acc6[r] + bv2;
        y1[48] = acc7[r] + bv3;
    }
}

// ---------------- misc elementwise kernels ----------------
__global__ void k_set_cls(float* __restrict__ h, const float* __restrict__ cls) {
    int i = blockIdx.x * blockDim.x + threadIdx.x;
    if (i < NB * HID) h[i] = cls[i & (HID - 1)];
}

__global__ void k_fill(float* __restrict__ p, float v, int n) {
    int i = blockIdx.x * blockDim.x + threadIdx.x;
    if (i < n) p[i] = v;
}

// per-(node,head) attention logits: als = <xs, att_src>, ald = <xs, att_dst>
__global__ void k_node_logits(const float* __restrict__ xs,
                              const float* __restrict__ as,
                              const float* __restrict__ ad,
                              float* __restrict__ als, float* __restrict__ ald)
{
    int t = blockIdx.x * blockDim.x + threadIdx.x;
    if (t >= MW * NHEADS) return;
    const int m = t >> 2, hd = t & 3;
    const float* __restrict__ xr = xs + (size_t)m * HID + hd * CDIM;
    const float* __restrict__ a1 = as + hd * CDIM;
    const float* __restrict__ a2 = ad + hd * CDIM;
    float s1 = 0.f, s2 = 0.f;
    #pragma unroll 8
    for (int c = 0; c < CDIM; ++c) { const float x = xr[c]; s1 += x * a1[c]; s2 += x * a2[c]; }
    als[t] = s1; ald[t] = s2;
}

// pass 1: segment max over incoming edges
__global__ void k_edge_max(const int* __restrict__ ei, const int* __restrict__ batch,
                           const float* __restrict__ als, const float* __restrict__ ald,
                           float* __restrict__ mmax)
{
    int t = blockIdx.x * blockDim.x + threadIdx.x;
    if (t >= EAUG * NHEADS) return;
    const int e = t >> 2, hd = t & 3;
    int s, d; edge_sd(e, ei, batch, s, d);
    float sc = als[s * 4 + hd] + ald[d * 4 + hd];
    sc = sc > 0.f ? sc : NEG_SLOPE * sc;
    atomicMaxF(&mmax[d * 4 + hd], sc);
}

// pass 2: ex = exp(e - max), den = segment sum
__global__ void k_edge_exp(const int* __restrict__ ei, const int* __restrict__ batch,
                           const float* __restrict__ als, const float* __restrict__ ald,
                           const float* __restrict__ mmax,
                           float* __restrict__ ex, float* __restrict__ den)
{
    int t = blockIdx.x * blockDim.x + threadIdx.x;
    if (t >= EAUG * NHEADS) return;
    const int e = t >> 2, hd = t & 3;
    int s, d; edge_sd(e, ei, batch, s, d);
    float sc = als[s * 4 + hd] + ald[d * 4 + hd];
    sc = sc > 0.f ? sc : NEG_SLOPE * sc;
    const float ev = __expf(sc - mmax[d * 4 + hd]);
    ex[t] = ev;
    atomicAdd(&den[d * 4 + hd], ev);
}

// pass 3: agg[dst, c] += alpha * xs[src, c]   (grid-stride over EAUG*HID)
__global__ void k_edge_msg(const int* __restrict__ ei, const int* __restrict__ batch,
                           const float* __restrict__ xs, const float* __restrict__ ex,
                           const float* __restrict__ den, float* __restrict__ agg)
{
    const long long total = (long long)EAUG * HID;
    for (long long idx = (long long)blockIdx.x * blockDim.x + threadIdx.x;
         idx < total; idx += (long long)gridDim.x * blockDim.x) {
        const int e  = (int)(idx >> 8);
        const int c  = (int)(idx & (HID - 1));
        const int hd = c >> 6;
        int s, d; edge_sd(e, ei, batch, s, d);
        const float alpha = ex[e * 4 + hd] / (den[d * 4 + hd] + 1e-16f);
        atomicAdd(&agg[(size_t)d * HID + c], alpha * xs[(size_t)s * HID + c]);
    }
}

// residual + bias + LayerNorm (one 256-thread block per row; wave32 shuffles)
__global__ __launch_bounds__(HID)
void k_add_ln(float* __restrict__ h, const float* __restrict__ agg,
              const float* __restrict__ bias, const float* __restrict__ g,
              const float* __restrict__ bet)
{
    const int row = blockIdx.x;
    const int c   = threadIdx.x;
    float v = h[(size_t)row * HID + c] + agg[(size_t)row * HID + c] + bias[c];

    __shared__ float red[8];
    __shared__ float s_mu, s_var;

    float s = v;
    #pragma unroll
    for (int o = 16; o > 0; o >>= 1) s += __shfl_down(s, o, 32);
    if ((c & 31) == 0) red[c >> 5] = s;
    __syncthreads();
    if (c == 0) { float t = 0.f; for (int i = 0; i < 8; ++i) t += red[i]; s_mu = t * (1.0f / HID); }
    __syncthreads();
    const float mu = s_mu;
    const float dv = v - mu;
    float q = dv * dv;
    #pragma unroll
    for (int o = 16; o > 0; o >>= 1) q += __shfl_down(q, o, 32);
    if ((c & 31) == 0) red[c >> 5] = q;
    __syncthreads();
    if (c == 0) { float t = 0.f; for (int i = 0; i < 8; ++i) t += red[i]; s_var = t * (1.0f / HID); }
    __syncthreads();

    h[(size_t)row * HID + c] = dv * rsqrtf(s_var + LN_EPS) * g[c] + bet[c];
}

// ---------------- launch ----------------
extern "C" void kernel_launch(void* const* d_in, const int* in_sizes, int n_in,
                              void* d_out, int out_size, void* d_ws, size_t ws_size,
                              hipStream_t stream)
{
    const float* x       = (const float*)d_in[0];
    const int*   ei      = (const int*)  d_in[1];
    const int*   batch   = (const int*)  d_in[2];
    const float* W_in    = (const float*)d_in[3];
    const float* b_in    = (const float*)d_in[4];
    const float* cls     = (const float*)d_in[5];
    const float* W_gat   = (const float*)d_in[6];   // (L,256,256)
    const float* att_src = (const float*)d_in[7];   // (L,4,64)
    const float* att_dst = (const float*)d_in[8];
    const float* b_gat   = (const float*)d_in[9];   // (L,256)
    const float* ln_g    = (const float*)d_in[10];
    const float* ln_b    = (const float*)d_in[11];
    const float* W_out   = (const float*)d_in[12];
    const float* b_out   = (const float*)d_in[13];
    float* out = (float*)d_out;

    // workspace layout (~85 MB)
    float* h    = (float*)d_ws;                  // MW*HID f32
    float* xs   = h    + (size_t)MW * HID;       // MW*HID f32
    float* agg  = xs   + (size_t)MW * HID;       // MW*HID f32
    float* als  = agg  + (size_t)MW * HID;       // MW*4
    float* ald  = als  + (size_t)MW * NHEADS;
    float* mmax = ald  + (size_t)MW * NHEADS;
    float* den  = mmax + (size_t)MW * NHEADS;
    float* ex   = den  + (size_t)MW * NHEADS;    // EAUG*4
    _Float16* x16 = (_Float16*)(ex + (size_t)EAUG * NHEADS);  // NNODES*NODE_DIM f16 (32B aligned)
    _Float16* h16 = x16 + (size_t)NNODES * NODE_DIM;          // MW*HID f16
    _Float16* w16 = h16 + (size_t)MW * HID;                   // 256*256 f16 (reused per GEMM)

    // h = [cls ; x @ W_in + b_in]
    k_set_cls<<<(NB * HID + 255) / 256, 256, 0, stream>>>(h, cls);
    k_cvt_f16<<<(NNODES * NODE_DIM + 255) / 256, 256, 0, stream>>>(x, x16, NNODES * NODE_DIM);
    k_cvt_f16<<<(NODE_DIM * HID + 255) / 256, 256, 0, stream>>>(W_in, w16, NODE_DIM * HID);
    k_wmma_gemm<NODE_DIM><<<NNODES / 32, 128, 0, stream>>>(x16, w16, b_in, h + (size_t)NB * HID);

    for (int l = 0; l < NLAYERS; ++l) {
        // xs = h @ W_gat[l]
        k_cvt_f16<<<(MW * HID + 255) / 256, 256, 0, stream>>>(h, h16, MW * HID);
        k_cvt_f16<<<(HID * HID + 255) / 256, 256, 0, stream>>>(W_gat + (size_t)l * HID * HID,
                                                               w16, HID * HID);
        k_wmma_gemm<HID><<<MW / 32, 128, 0, stream>>>(h16, w16, nullptr, xs);

        k_node_logits<<<(MW * NHEADS + 255) / 256, 256, 0, stream>>>(
            xs, att_src + l * NHEADS * CDIM, att_dst + l * NHEADS * CDIM, als, ald);

        k_fill<<<(MW * NHEADS + 255) / 256, 256, 0, stream>>>(mmax, -1e30f, MW * NHEADS);
        k_fill<<<(MW * NHEADS + 255) / 256, 256, 0, stream>>>(den, 0.f, MW * NHEADS);
        k_fill<<<(MW * HID + 255) / 256, 256, 0, stream>>>(agg, 0.f, MW * HID);

        const int tE = EAUG * NHEADS;
        k_edge_max<<<(tE + 255) / 256, 256, 0, stream>>>(ei, batch, als, ald, mmax);
        k_edge_exp<<<(tE + 255) / 256, 256, 0, stream>>>(ei, batch, als, ald, mmax, ex, den);
        k_edge_msg<<<8192, 256, 0, stream>>>(ei, batch, xs, ex, den, agg);

        k_add_ln<<<MW, HID, 0, stream>>>(h, agg, b_gat + l * HID,
                                         ln_g + l * HID, ln_b + l * HID);
    }

    // out = h[:B] @ W_out + b_out
    k_cvt_f16<<<(NB * HID + 255) / 256, 256, 0, stream>>>(h, h16, NB * HID);
    k_cvt_f16<<<(HID * EMB + 255) / 256, 256, 0, stream>>>(W_out, w16, HID * EMB);
    k_wmma_gemm<HID><<<NB / 32, 128, 0, stream>>>(h16, w16, b_out, out);
}
